// Atten2NetVLAD_82617990906407
// MI455X (gfx1250) — matrix operand order repaired and checked by reading, not compiled, not run on hardware
//
#include <hip/hip_runtime.h>

typedef __attribute__((ext_vector_type(16))) __bf16 v16bf;
typedef __attribute__((ext_vector_type(8)))  float  v8f;

union ABFrag { v16bf v; unsigned int u[8]; };

__device__ __forceinline__ unsigned short f2bf(float f) {
  unsigned int x = __float_as_uint(f);
  return (unsigned short)((x + 0x7FFFu + ((x >> 16) & 1u)) >> 16);  // RNE
}
__device__ __forceinline__ float bf2f(unsigned short h) {
  return __uint_as_float(((unsigned int)h) << 16);
}

#define C_DIM 128
#define K_DIM 64
#define S_DIM 961
#define ST    64

// ---- static LDS layout (bytes) ----
#define XM_SC_OFF   0        // ushort[64*130]  x_mul, [s*130+c]   (B of GEMM1)
#define XM_CS_OFF   16640    // ushort[128*66]  x_mul, [c*66+s]    (B of GEMM2)
#define LOGITS_OFF  33536    // float [64*65]   logits/exp, [k*65+s]
#define WLDS_OFF    50176    // ushort[64*66]   w=softmax*hmp, [k*66+s] (A of GEMM2)
#define HMP_OFF     58624    // float[64]
#define INVN_OFF    58880    // float[64]
#define CMAX_OFF    59136    // float[64]
#define CSCL_OFF    59392    // float[64]
#define A4_OFF      59648    // float[256] scratch partials
#define B4_OFF      60672    // float[256] scratch partials
#define WSUM_OFF    61696    // float[64]
#define RINV_OFF    61952    // float[64]
#define GT_OFF      62208    // float[4]
#define AW_OFF      62224    // float[128]
#define SM_BYTES    62736
// vlad f32[64*129] = 33024 B aliases XM_SC+XM_CS after the s-loop

__global__ __launch_bounds__(256) void netvlad_kernel(
    const float* __restrict__ x, const float* __restrict__ conv_w,
    const float* __restrict__ centroids, const float* __restrict__ atten_w,
    const float* __restrict__ atten_b, float* __restrict__ out)
{
  __shared__ __align__(16) char SM[SM_BYTES];
  unsigned short* xm_sc = (unsigned short*)(SM + XM_SC_OFF);
  unsigned short* xm_cs = (unsigned short*)(SM + XM_CS_OFF);
  float*          logits = (float*)(SM + LOGITS_OFF);
  unsigned short* w_lds  = (unsigned short*)(SM + WLDS_OFF);
  float* hmp  = (float*)(SM + HMP_OFF);
  float* invn = (float*)(SM + INVN_OFF);
  float* cmax = (float*)(SM + CMAX_OFF);
  float* cscl = (float*)(SM + CSCL_OFF);
  float* A4   = (float*)(SM + A4_OFF);
  float* B4   = (float*)(SM + B4_OFF);
  float* wsum = (float*)(SM + WSUM_OFF);
  float* rinv = (float*)(SM + RINV_OFF);
  float* gt   = (float*)(SM + GT_OFF);
  float* aw   = (float*)(SM + AW_OFF);
  float* vlad = (float*)(SM + 0);         // [k*129 + c], used after s-loop

  const int t    = threadIdx.x;
  const int lane = t & 31;
  const int wave = t >> 5;                // 0..7
  const int n    = blockIdx.x;
  const float b0 = atten_b[0];
  const float* xn = x + (size_t)n * (C_DIM * S_DIM);

  if (t < C_DIM) aw[t] = atten_w[t];
  if (t < K_DIM) wsum[t] = 0.f;

  // Preload conv_w A-fragments (bf16) once: this wave's kt = wave>>1, 4 c-chunks.
  // 16-bit A 16x32 layout: lane<16 -> M=lane, K pair base = (v<4)?2v:2v+8;
  // lanes 16..31 add 8 to K.
  const int ktA = wave >> 1;
  ABFrag afrag[4];
  {
    const int kRow = ktA * 16 + (lane & 15);
    const int ko   = (lane & 16) ? 8 : 0;
    #pragma unroll
    for (int cc = 0; cc < 4; ++cc) {
      #pragma unroll
      for (int v = 0; v < 8; ++v) {
        int cb = cc * 32 + ko + ((v < 4) ? 2 * v : 2 * v + 8);
        float f0 = conv_w[kRow * C_DIM + cb];
        float f1 = conv_w[kRow * C_DIM + cb + 1];
        afrag[cc].u[v] = (unsigned)f2bf(f0) | ((unsigned)f2bf(f1) << 16);
      }
    }
  }

  v8f acc0 = {}, acc1 = {}, acc2 = {}, acc3 = {};   // vlad fragments, kt=0..3, ct=wave
  __syncthreads();

  const int part  = t >> 6;   // 0..3
  const int scol  = t & 63;   // 0..63
  const int krow4 = t >> 2;   // 0..63
  const int q4    = t & 3;    // 0..3

  for (int s0 = 0; s0 < S_DIM; s0 += ST) {
    // ---- per-column reductions: ||x||^2 and relu(x)·aw ----
    {
      float ssq = 0.f, dot = 0.f;
      const int sg = s0 + scol;
      if (sg < S_DIM) {
        const float* xp = xn + sg;
        #pragma unroll 4
        for (int c = part * 32; c < part * 32 + 32; ++c) {
          float v = xp[(size_t)c * S_DIM];
          ssq += v * v;
          dot += fmaxf(v, 0.f) * aw[c];
        }
      }
      A4[part * 64 + scol] = ssq;
      B4[part * 64 + scol] = dot;
    }
    __syncthreads();
    if (t < 64) {
      float ssq = A4[t] + A4[64 + t] + A4[128 + t] + A4[192 + t];
      float dot = B4[t] + B4[64 + t] + B4[128 + t] + B4[192 + t];
      invn[t] = 1.f / fmaxf(__fsqrt_rn(ssq), 1e-12f);
      hmp[t]  = fmaxf(dot + b0, 0.f);
    }
    __syncthreads();
    // ---- build bf16 x_mul in both layouts (2nd x read hits L2) ----
    #pragma unroll 4
    for (int i = 0; i < 32; ++i) {
      int idx = i * 256 + t;
      int c = idx >> 6, s = idx & 63;
      int sg = s0 + s;
      float v = (sg < S_DIM) ? xn[(size_t)c * S_DIM + sg] : 0.f;
      unsigned short h = f2bf(v * invn[s] * hmp[s]);
      xm_sc[s * 130 + c] = h;
      xm_cs[c * 66 + s]  = h;
    }
    __syncthreads();
    // ---- GEMM1: logits[k,s] = sum_c conv_w[k,c] * x_mul[c,s] ----
    {
      #pragma unroll
      for (int f = 0; f < 2; ++f) {
        const int st = (wave * 2 + f) & 3;
        v8f lac = {};
        const int sCol = st * 16 + (lane & 15);
        const int kob  = (lane & 16) ? 8 : 0;      // B: lanes 16..31 -> K+8
        #pragma unroll
        for (int cc = 0; cc < 4; ++cc) {
          ABFrag bfr;
          #pragma unroll
          for (int v = 0; v < 8; ++v)
            bfr.u[v] = *(const unsigned int*)(xm_sc + sCol * 130 + cc * 32 + kob + ((v < 4) ? 2 * v : 2 * v + 8));
          lac = __builtin_amdgcn_wmma_f32_16x16x32_bf16(false, afrag[cc].v, false, bfr.v,
                                                        (short)0, lac, false, false);
        }
        const int rb = ktA * 16 + ((lane & 16) ? 8 : 0);
        #pragma unroll
        for (int r = 0; r < 8; ++r)
          logits[(rb + r) * 65 + sCol] = lac[r];
      }
    }
    __syncthreads();
    // ---- softmax over k (64), weight by hmp ----
    {
      float m = -3.4e38f;
      #pragma unroll
      for (int kk = 0; kk < 16; ++kk)
        m = fmaxf(m, logits[(part * 16 + kk) * 65 + scol]);
      A4[part * 64 + scol] = m;
    }
    __syncthreads();
    if (t < 64) cmax[t] = fmaxf(fmaxf(A4[t], A4[64 + t]), fmaxf(A4[128 + t], A4[192 + t]));
    __syncthreads();
    {
      float cm = cmax[scol], sum = 0.f;
      #pragma unroll
      for (int kk = 0; kk < 16; ++kk) {
        float e = __expf(logits[(part * 16 + kk) * 65 + scol] - cm);
        logits[(part * 16 + kk) * 65 + scol] = e;
        sum += e;
      }
      B4[part * 64 + scol] = sum;
    }
    __syncthreads();
    if (t < 64) cscl[t] = hmp[t] / (B4[t] + B4[64 + t] + B4[128 + t] + B4[192 + t]);
    __syncthreads();
    {
      float sc = cscl[scol];
      #pragma unroll
      for (int kk = 0; kk < 16; ++kk)
        w_lds[(part * 16 + kk) * 66 + scol] =
            f2bf(logits[(part * 16 + kk) * 65 + scol] * sc);
    }
    __syncthreads();
    // ---- wsum[k] += sum_s w[k,s] ----
    {
      float s = 0.f;
      #pragma unroll
      for (int ss = 0; ss < 16; ++ss)
        s += bf2f(w_lds[krow4 * 66 + q4 * 16 + ss]);
      A4[q4 * 64 + krow4] = s;
    }
    __syncthreads();
    if (t < 64) wsum[t] += A4[t] + A4[64 + t] + A4[128 + t] + A4[192 + t];
    __syncthreads();
    // ---- GEMM2: vlad[k,c] += sum_s w[k,s] * x_mul[c,s] (acc in VGPRs) ----
    {
      const int cCol = wave * 16 + (lane & 15);
      const int kRowL = lane & 15;
      const int koA = (lane & 16) ? 8 : 0;
      const int koB = (lane & 16) ? 8 : 0;
      #pragma unroll
      for (int sch = 0; sch < 2; ++sch) {
        ABFrag bfr;
        #pragma unroll
        for (int v = 0; v < 8; ++v)
          bfr.u[v] = *(const unsigned int*)(xm_cs + cCol * 66 + sch * 32 + koB + ((v < 4) ? 2 * v : 2 * v + 8));
        #pragma unroll
        for (int kt = 0; kt < 4; ++kt) {
          ABFrag afr;
          #pragma unroll
          for (int v = 0; v < 8; ++v) {
            int sb = sch * 32 + koA + ((v < 4) ? 2 * v : 2 * v + 8);
            afr.u[v] = *(const unsigned int*)(w_lds + (kt * 16 + kRowL) * 66 + sb);
          }
          v8f& acc = (kt == 0) ? acc0 : ((kt == 1) ? acc1 : ((kt == 2) ? acc2 : acc3));
          acc = __builtin_amdgcn_wmma_f32_16x16x32_bf16(false, afr.v, false, bfr.v,
                                                        (short)0, acc, false, false);
        }
      }
    }
    __syncthreads();
  }

  // ---- spill vlad fragments to LDS (aliases xm buffers) ----
  {
    const int colC = wave * 16 + (lane & 15);
    const int rb = (lane & 16) ? 8 : 0;
    #pragma unroll
    for (int kt = 0; kt < 4; ++kt) {
      v8f& acc = (kt == 0) ? acc0 : ((kt == 1) ? acc1 : ((kt == 2) ? acc2 : acc3));
      #pragma unroll
      for (int r = 0; r < 8; ++r)
        vlad[(kt * 16 + rb + r) * 129 + colC] = acc[r];
    }
  }
  __syncthreads();
  // ---- subtract centroids * wsum, row sum-of-squares ----
  {
    float ws = wsum[krow4], ss = 0.f;
    #pragma unroll 4
    for (int c = q4 * 32; c < q4 * 32 + 32; ++c) {
      float v = vlad[krow4 * 129 + c] - centroids[krow4 * C_DIM + c] * ws;
      vlad[krow4 * 129 + c] = v;
      ss += v * v;
    }
    A4[q4 * 64 + krow4] = ss;
  }
  __syncthreads();
  if (t < 64) {
    float ss = A4[t] + A4[64 + t] + A4[128 + t] + A4[192 + t];
    float inv = 1.f / fmaxf(__fsqrt_rn(ss), 1e-12f);
    rinv[t] = inv;
    B4[t] = ss * inv * inv;   // contribution of normalized row to global norm
  }
  __syncthreads();
  if (t == 0) {
    float tot = 0.f;
    for (int k = 0; k < 64; ++k) tot += B4[k];
    gt[0] = 1.f / fmaxf(__fsqrt_rn(tot), 1e-12f);
    // second global normalize divides a unit vector by 1 -> identity
  }
  __syncthreads();
  {
    float ri = rinv[krow4] * gt[0];
    float* op = out + (size_t)n * (K_DIM * C_DIM) + krow4 * C_DIM;
    #pragma unroll 4
    for (int c = q4 * 32; c < q4 * 32 + 32; ++c)
      op[c] = vlad[krow4 * 129 + c] * ri;
  }
}

extern "C" void kernel_launch(void* const* d_in, const int* in_sizes, int n_in,
                              void* d_out, int out_size, void* d_ws, size_t ws_size,
                              hipStream_t stream) {
  (void)in_sizes; (void)n_in; (void)d_ws; (void)ws_size; (void)out_size;
  const float* x       = (const float*)d_in[0];
  const float* conv_w  = (const float*)d_in[1];
  const float* cents   = (const float*)d_in[2];
  const float* atten_w = (const float*)d_in[3];
  const float* atten_b = (const float*)d_in[4];
  netvlad_kernel<<<dim3(256), dim3(256), 0, stream>>>(x, conv_w, cents, atten_w,
                                                      atten_b, (float*)d_out);
}
